// MultiHeadAttentionBlock_57105885167707
// MI455X (gfx1250) — matrix operand order repaired and checked by reading, not compiled, not run on hardware
//
#include <hip/hip_runtime.h>
#include <cstdint>

// ---------------------------------------------------------------------------
// MultiHeadAttentionBlock for MI455X (gfx1250, wave32, WMMA bf16 path)
// B=2, S=2048, D=1024, H=16, AD=64.
// Pipeline: fp32->bf16 convert (weights pre-transposed) ->
//           bf16 WMMA GEMM with async global->LDS double-buffered staging ->
//           flash attention (all-bf16 vector fragment loads) ->
//           bf16 WMMA GEMM -> fp32 out.
// ---------------------------------------------------------------------------

typedef __attribute__((ext_vector_type(16))) __bf16   v16bf;
typedef __attribute__((ext_vector_type(8)))  float    v8f;
typedef __attribute__((ext_vector_type(8)))  unsigned v8u;
typedef unsigned short ushort_t;

union BFrag { v8u u; v16bf b; uint4 q[2]; };   // 16 bf16 (A or B fragment)
union FFrag { v8f v; float f[8]; };            // 8 f32 (C/D fragment)

__device__ __forceinline__ ushort_t f2bf(float x) {
  unsigned u = __float_as_uint(x);
  u = u + 0x7FFFu + ((u >> 16) & 1u);          // round-to-nearest-even
  return (ushort_t)(u >> 16);
}
__device__ __forceinline__ unsigned f2bf2(float lo, float hi) {
  return (unsigned)f2bf(lo) | ((unsigned)f2bf(hi) << 16);
}
__device__ __forceinline__ v8f wmma_bf16(v16bf a, v16bf b, v8f c) {
  return __builtin_amdgcn_wmma_f32_16x16x32_bf16(false, a, false, b,
                                                 (short)0, c, false, false);
}

// ---- CDNA5 async global->LDS copy (16B per lane), tracked by ASYNCcnt ----
__device__ __forceinline__ void async_cp_b128(unsigned ldsAddr, const void* g) {
  asm volatile("global_load_async_to_lds_b128 %0, %1, off"
               :: "v"(ldsAddr), "v"(g) : "memory");
}
__device__ __forceinline__ void wait_async0() {
  asm volatile("s_wait_asynccnt 0" ::: "memory");
}

// ---------------------------------------------------------------------------
// Elementwise fp32 -> bf16 (8 elements / thread), one-shot memory-bound pass.
// ---------------------------------------------------------------------------
__global__ __launch_bounds__(256)
void cvt_f32_bf16(const float4* __restrict__ in, uint4* __restrict__ out, int n8) {
  int idx = blockIdx.x * blockDim.x + threadIdx.x;
  if (idx < n8) {
    float4 a = in[2 * idx], b = in[2 * idx + 1];
    uint4 o;
    o.x = f2bf2(a.x, a.y);
    o.y = f2bf2(a.z, a.w);
    o.z = f2bf2(b.x, b.y);
    o.w = f2bf2(b.z, b.w);
    out[idx] = o;
  }
}

// ---------------------------------------------------------------------------
// fp32 [R,C] -> bf16 transposed [C,R], 32x32 tiles through LDS (coalesced).
// ---------------------------------------------------------------------------
__global__ __launch_bounds__(256)
void cvt_f32_bf16_T(const float* __restrict__ in, ushort_t* __restrict__ out,
                    int R, int C) {
  __shared__ float t[32][33];
  int r0 = blockIdx.y * 32, c0 = blockIdx.x * 32;
  int tx = threadIdx.x & 31, ty = threadIdx.x >> 5;   // ty = 0..7
#pragma unroll
  for (int rr = ty; rr < 32; rr += 8)
    t[rr][tx] = in[(size_t)(r0 + rr) * C + c0 + tx];
  __syncthreads();
#pragma unroll
  for (int cc = ty; cc < 32; cc += 8)
    out[(size_t)(c0 + cc) * R + r0 + tx] = f2bf(t[tx][cc]);
}

// ---------------------------------------------------------------------------
// bf16 GEMM: C[M,N] = A[M,K] @ Wt[N,K]^T (+ bias).  Block 128 thr = 4 waves.
// Block tile 128(M) x 64(N), K step 32.  Wave w: rows 32w..32w+31 -> 2 A frags,
// 4 B frags shared across both -> 8 WMMAs / wave / K-step.
// Staging: double-buffered async global->LDS b128 copies (ASYNCcnt).
// Epilogue modes:
//   0: fp32 [M,N] + bias              (final projection)
//   1: bf16 head-major  [b,h,s,64]    (Q, K)
//   2: bf16 head-major-T [b,h,64,s]   (V)
// ---------------------------------------------------------------------------
#define MODE_F32_BIAS   0
#define MODE_BF16_HEAD  1
#define MODE_BF16_HEADT 2

template <int MODE>
__global__ __launch_bounds__(128)
void gemm_bf16(const ushort_t* __restrict__ A, const ushort_t* __restrict__ Wt,
               const float* __restrict__ bias, void* __restrict__ Cout,
               int M, int N, int K) {
  __shared__ alignas(16) ushort_t As[2][128][32];   // [m][k]
  __shared__ alignas(16) ushort_t Bs[2][64][32];    // [n][k] (Wt row-major)

  const int tid  = threadIdx.x;
  const int wave = tid >> 5;
  const int lane = tid & 31;
  const int lo   = lane & 15;
  const int hi   = lane >> 4;
  const int mbase = blockIdx.y * 128;
  const int nbase = blockIdx.x * 64;

  auto stageA = [&](int buf, int k0) {
#pragma unroll
    for (int j = 0; j < 4; j++) {
      int linear = tid + 128 * j;
      int r = linear >> 2, c8 = (linear & 3) * 8;
      async_cp_b128((unsigned)(uintptr_t)&As[buf][r][c8],
                    &A[(size_t)(mbase + r) * K + k0 + c8]);
    }
  };
  auto stageB = [&](int buf, int k0) {
#pragma unroll
    for (int j = 0; j < 2; j++) {
      int linear = tid + 128 * j;
      int r = linear >> 2, c8 = (linear & 3) * 8;
      async_cp_b128((unsigned)(uintptr_t)&Bs[buf][r][c8],
                    &Wt[(size_t)(nbase + r) * K + k0 + c8]);
    }
  };

  float acc[2][4][8];
#pragma unroll
  for (int s = 0; s < 2; s++)
#pragma unroll
    for (int f = 0; f < 4; f++)
#pragma unroll
      for (int i = 0; i < 8; i++) acc[s][f][i] = 0.0f;

  // prologue: fill buffer 0
  stageA(0, 0);
  stageB(0, 0);
  wait_async0();
  __syncthreads();

  int p = 0;
  for (int k0 = 0; k0 < K; k0 += 32) {
    const bool more = (k0 + 32) < K;
    if (more) {               // kick off next tile while computing this one
      stageA(p ^ 1, k0 + 32);
      stageB(p ^ 1, k0 + 32);
    }

    // ---- fragments from LDS buffer p ----
    BFrag af[2], bf[4];
#pragma unroll
    for (int s = 0; s < 2; s++) {
      const uint4* ar = (const uint4*)&As[p][wave * 32 + s * 16 + lo][0];
      af[s].q[0] = ar[hi];          // K = 8*hi + 0..7
      af[s].q[1] = ar[2 + hi];      // K = 16 + 8*hi + 0..7
    }
#pragma unroll
    for (int f = 0; f < 4; f++) {
      const uint4* br = (const uint4*)&Bs[p][f * 16 + lo][0];
      bf[f].q[0] = br[hi];          // K = 8*hi + 0..7
      bf[f].q[1] = br[2 + hi];      // K = 16 + 8*hi + 0..7
    }
    // ---- 8 WMMAs ----
#pragma unroll
    for (int s = 0; s < 2; s++)
#pragma unroll
      for (int f = 0; f < 4; f++) {
        FFrag c;
#pragma unroll
        for (int i = 0; i < 8; i++) c.f[i] = acc[s][f][i];
        c.v = wmma_bf16(af[s].b, bf[f].b, c.v);
#pragma unroll
        for (int i = 0; i < 8; i++) acc[s][f][i] = c.f[i];
      }

    if (more) wait_async0();   // our async copies into buffer p^1 done
    __syncthreads();           // everyone done reading p / writing p^1
    p ^= 1;
  }

  // ---- epilogue: C layout row = i + 8*hi (in 16-row frag), col = 16f + lo ----
#pragma unroll
  for (int s = 0; s < 2; s++)
#pragma unroll
    for (int f = 0; f < 4; f++) {
      int col = nbase + f * 16 + lo;
      float bv = bias[col];
#pragma unroll
      for (int i = 0; i < 8; i++) {
        int row = mbase + wave * 32 + s * 16 + i + 8 * hi;
        float val = acc[s][f][i] + bv;
        if constexpr (MODE == MODE_F32_BIAS) {
          ((float*)Cout)[(size_t)row * N + col] = val;
        } else {
          int bb = row >> 11, ss = row & 2047;   // S = 2048
          int hh = col >> 6,  ad = col & 63;     // AD = 64
          size_t addr;
          if constexpr (MODE == MODE_BF16_HEAD)
            addr = (((size_t)(bb * 16 + hh) * 2048) + ss) * 64 + ad;
          else
            addr = (((size_t)(bb * 16 + hh) * 64) + ad) * 2048 + ss;
          ((ushort_t*)Cout)[addr] = f2bf(val);
        }
      }
    }
}

// ---------------------------------------------------------------------------
// Flash attention, causal.  One wave per 16-row query tile of one (b,h).
// All operands pre-converted bf16:  Qh/Kh = [b,h,s,64], Vt = [b,h,64,S].
// Key tiles of 32: QK^T = 4 WMMAs, PV = 4 WMMAs per tile.
// ---------------------------------------------------------------------------
__global__ __launch_bounds__(128)
void flash_attn_causal(const ushort_t* __restrict__ Qh,
                       const ushort_t* __restrict__ Kh,
                       const ushort_t* __restrict__ Vt,
                       ushort_t* __restrict__ At) {
  constexpr int S = 2048;
  __shared__ alignas(16) ushort_t Pst[4][16 * 32];  // per-wave staging

  const int tid  = threadIdx.x;
  const int wave = tid >> 5;
  const int lane = tid & 31;
  const int lo   = lane & 15;
  const int hi   = lane >> 4;
  const int bh = blockIdx.y;                  // b*16 + h
  const int h  = bh & 15;
  const int qt = blockIdx.x * 4 + wave;       // 0..127

  const ushort_t* Qb = Qh + (size_t)bh * S * 64;
  const ushort_t* Kb = Kh + (size_t)bh * S * 64;
  const ushort_t* Vb = Vt + (size_t)bh * 64 * S;

  // ---- Q fragments (K-dim 64 -> 2 A frags), straight b128 loads ----
  BFrag qa[2];
  {
    const uint4* qr = (const uint4*)(Qb + (size_t)(qt * 16 + lo) * 64);
#pragma unroll
    for (int halfk = 0; halfk < 2; halfk++) {
      qa[halfk].q[0] = qr[halfk * 4 + hi];       // dims halfk*32 + 8*hi + 0..7
      qa[halfk].q[1] = qr[halfk * 4 + 2 + hi];   // dims halfk*32 + 16 + 8*hi ..
    }
  }

  float mrow[8], lrow[8], oacc[4][8];
#pragma unroll
  for (int i = 0; i < 8; i++) { mrow[i] = -3.0e38f; lrow[i] = 0.0f; }
#pragma unroll
  for (int f = 0; f < 4; f++)
#pragma unroll
    for (int i = 0; i < 8; i++) oacc[f][i] = 0.0f;

  const int lastkt = (qt * 16 + 15) >> 5;        // inclusive

  for (int kt = 0; kt <= lastkt; kt++) {
    if (kt < lastkt) {   // prefetch next K/V tiles
      __builtin_prefetch(Kb + (size_t)(kt * 32 + 32 + lane) * 64, 0, 3);
      __builtin_prefetch(Vb + (size_t)(lane * 2) * S + kt * 32 + 32, 0, 3);
    }
    // ---- scores: two 16x16 C frags; B frag = K row (dims contiguous) ----
    FFrag sc[2];
#pragma unroll
    for (int s = 0; s < 2; s++) {
#pragma unroll
      for (int i = 0; i < 8; i++) sc[s].f[i] = 0.0f;
      int key = kt * 32 + s * 16 + lo;
      const uint4* kr = (const uint4*)(Kb + (size_t)key * 64);
#pragma unroll
      for (int halfk = 0; halfk < 2; halfk++) {
        BFrag kf;
        kf.q[0] = kr[halfk * 4 + hi];            // dims halfk*32 + 8*hi + 0..7
        kf.q[1] = kr[halfk * 4 + 2 + hi];        // dims halfk*32 + 16 + 8*hi ..
        sc[s].v = wmma_bf16(qa[halfk].b, kf.b, sc[s].v);
      }
      // scale 1/sqrt(64)
#pragma unroll
      for (int i = 0; i < 8; i++) sc[s].f[i] *= 0.125f;
    }

    // ---- causal mask on diagonal-overlapping tiles ----
    if (kt * 32 + 31 > qt * 16 + 15) {
#pragma unroll
      for (int s = 0; s < 2; s++) {
        int kk = kt * 32 + s * 16 + lo;
#pragma unroll
        for (int i = 0; i < 8; i++) {
          int qq = qt * 16 + i + 8 * hi;
          if (kk > qq) sc[s].f[i] = -1.0e30f;
        }
      }
    }

    // ---- online softmax; row m = i + 8*hi lives in one 16-lane half ----
#pragma unroll
    for (int i = 0; i < 8; i++) {
      float rm = fmaxf(sc[0].f[i], sc[1].f[i]);
#pragma unroll
      for (int off = 1; off < 16; off <<= 1)
        rm = fmaxf(rm, __shfl_xor(rm, off, 32));
      float mn = fmaxf(mrow[i], rm);
      float al = __expf(mrow[i] - mn);
      float p0 = __expf(sc[0].f[i] - mn);
      float p1 = __expf(sc[1].f[i] - mn);
      float rs = p0 + p1;
#pragma unroll
      for (int off = 1; off < 16; off <<= 1)
        rs += __shfl_xor(rs, off, 32);
      lrow[i] = lrow[i] * al + rs;
      mrow[i] = mn;
#pragma unroll
      for (int f = 0; f < 4; f++) oacc[f][i] *= al;
      int m = i + 8 * hi;                        // stage P in C layout
      Pst[wave][m * 32 + lo]      = f2bf(p0);
      Pst[wave][m * 32 + 16 + lo] = f2bf(p1);
    }

    // ---- P as A fragment (16x32 over keys); LDS is in-order per wave ----
    BFrag pf;
    {
      const uint4* pr = (const uint4*)&Pst[wave][lo * 32];
      pf.q[0] = pr[hi];
      pf.q[1] = pr[2 + hi];
    }

    // ---- O += P @ V ; V frag rows contiguous along keys (Vt layout) ----
#pragma unroll
    for (int f = 0; f < 4; f++) {
      BFrag vf;
      const uint4* vr = (const uint4*)(Vb + (size_t)(f * 16 + lo) * S + kt * 32);
      vf.q[0] = vr[hi];                          // keys 8*hi + 0..7
      vf.q[1] = vr[2 + hi];                      // keys 16 + 8*hi + 0..7
      FFrag c;
#pragma unroll
      for (int i = 0; i < 8; i++) c.f[i] = oacc[f][i];
      c.v = wmma_bf16(pf.b, vf.b, c.v);
#pragma unroll
      for (int i = 0; i < 8; i++) oacc[f][i] = c.f[i];
    }
  }

  // ---- finalize: /= rowsum, store bf16 row-major [b*S+s, h*64+d] ----
#pragma unroll
  for (int i = 0; i < 8; i++) {
    float inv = 1.0f / lrow[i];
    int row = (bh >> 4) * S + qt * 16 + i + 8 * hi;
    ushort_t* op = At + (size_t)row * 1024 + h * 64;
#pragma unroll
    for (int f = 0; f < 4; f++) op[f * 16 + lo] = f2bf(oacc[f][i] * inv);
  }
}

// ---------------------------------------------------------------------------
// Launch.  in_sizes order: q,k,v,mask,Wq,bq,Wk,bk,Wv,bv,Wp,bp
// ---------------------------------------------------------------------------
extern "C" void kernel_launch(void* const* d_in, const int* in_sizes, int n_in,
                              void* d_out, int out_size, void* d_ws, size_t ws_size,
                              hipStream_t stream) {
  (void)in_sizes; (void)n_in; (void)out_size; (void)ws_size;
  constexpr int M = 4096, NN = 1024, KK = 1024;     // all GEMMs 4096x1024x1024
  constexpr size_t ACT = (size_t)M * NN;            // 4 Mi elements
  constexpr size_t WGT = (size_t)KK * NN;           // 1 Mi elements

  const float* q  = (const float*)d_in[0];
  const float* k  = (const float*)d_in[1];
  const float* v  = (const float*)d_in[2];
  const float* Wq = (const float*)d_in[4];
  const float* bq = (const float*)d_in[5];
  const float* Wk = (const float*)d_in[6];
  const float* bk = (const float*)d_in[7];
  const float* Wv = (const float*)d_in[8];
  const float* bv = (const float*)d_in[9];
  const float* Wp = (const float*)d_in[10];
  const float* bp = (const float*)d_in[11];
  float* out = (float*)d_out;

  ushort_t* wsp = (ushort_t*)d_ws;
  ushort_t* qb  = wsp;              // bf16 copies of inputs
  ushort_t* kb  = qb  + ACT;
  ushort_t* vb  = kb  + ACT;
  ushort_t* wqt = vb  + ACT;        // bf16 W^T  [N,K]
  ushort_t* wkt = wqt + WGT;
  ushort_t* wvt = wkt + WGT;
  ushort_t* wpt = wvt + WGT;
  ushort_t* Qh  = wpt + WGT;        // projected, head-major bf16
  ushort_t* Khd = Qh  + ACT;
  ushort_t* Vtb = Khd + ACT;        // transposed [b,h,64,S]
  ushort_t* At  = Vtb + ACT;        // attention out, row-major bf16

  // 1) one-shot fp32 -> bf16 conversions; weights transposed to [N,K]
  cvt_f32_bf16<<<(int)(ACT / 8 / 256), 256, 0, stream>>>((const float4*)q, (uint4*)qb, (int)(ACT / 8));
  cvt_f32_bf16<<<(int)(ACT / 8 / 256), 256, 0, stream>>>((const float4*)k, (uint4*)kb, (int)(ACT / 8));
  cvt_f32_bf16<<<(int)(ACT / 8 / 256), 256, 0, stream>>>((const float4*)v, (uint4*)vb, (int)(ACT / 8));
  dim3 gT(NN / 32, KK / 32);        // (32, 32)
  cvt_f32_bf16_T<<<gT, 256, 0, stream>>>(Wq, wqt, KK, NN);
  cvt_f32_bf16_T<<<gT, 256, 0, stream>>>(Wk, wkt, KK, NN);
  cvt_f32_bf16_T<<<gT, 256, 0, stream>>>(Wv, wvt, KK, NN);
  cvt_f32_bf16_T<<<gT, 256, 0, stream>>>(Wp, wpt, KK, NN);

  // 2) projections (bf16 WMMA, async-staged), head-major outputs; V transposed
  dim3 gG(NN / 64, M / 128);        // (16, 32)
  gemm_bf16<MODE_BF16_HEAD ><<<gG, 128, 0, stream>>>(qb, wqt, bq, Qh,  M, NN, KK);
  gemm_bf16<MODE_BF16_HEAD ><<<gG, 128, 0, stream>>>(kb, wkt, bk, Khd, M, NN, KK);
  gemm_bf16<MODE_BF16_HEADT><<<gG, 128, 0, stream>>>(vb, wvt, bv, Vtb, M, NN, KK);

  // 3) flash attention
  flash_attn_causal<<<dim3(32, 32), 128, 0, stream>>>(Qh, Khd, Vtb, At);

  // 4) output projection -> fp32 + bias
  gemm_bf16<MODE_F32_BIAS><<<gG, 128, 0, stream>>>(At, wpt, bp, out, M, NN, KK);
}
